// VSACoding_10806137717263
// MI455X (gfx1250) — compile-verified, matrix operand order checked by reading, not run on hardware
//
#include <hip/hip_runtime.h>

typedef __attribute__((ext_vector_type(16))) _Float16 v16h;
typedef __attribute__((ext_vector_type(8)))  _Float16 v8h;
typedef __attribute__((ext_vector_type(8)))  float    v8f;

#define BI 64
#define BC 128
#define NR 36
#define NW 50
#define NR_PAD 48        // padded region rows (3 M tiles)
#define NW_PAD 64        // padded word rows (4 N tiles)
#define EMB 512
#define LDS_STRIDE 520   // halves per row: 512 + 8 pad -> 1040B, conflict-free b128 frag loads
#define S_STRIDE 66      // floats per row of score/P buffer
#define A_STRIDE 72      // halves per row of attn buffer
#define GDIM 64          // padded Gram dimension
#define TEMP_INV 10.0f
#define EPS 1e-8f

// A/B fragment for v_wmma_f32_16x16x32_f16 from a row-major f16 array.
// lane L: row/col index = tileBase + (L&15), kbase = 32*kstep + (L>>4)*8.
// elements 0..7  = [kbase .. kbase+7], elements 8..15 = [kbase+16 .. kbase+23].
// Rows are zero-padded in LDS, so the load is unconditional (no EXEC churn).
__device__ __forceinline__ v16h frag_load(const _Float16* base, int row, int kbase,
                                          int stride) {
  union { v16h v; v8h h[2]; } u;
  const _Float16* p = base + row * stride + kbase;
  u.h[0] = *(const v8h*)(p);
  u.h[1] = *(const v8h*)(p + 16);
  return u.v;
}

// Cooperative load of nrows x 512 f32 rows from global into f16 LDS
// (stride LDS_STRIDE), zero-filling padded rows [nrows, padrows).
__device__ __forceinline__ void load_f32_rows_to_f16_lds(const float* __restrict__ src,
                                                         _Float16* dst, int nrows,
                                                         int padrows, int tid,
                                                         int nthreads) {
  const int total4 = padrows * (EMB / 4);
  const float4* s4 = (const float4*)src;
  for (int idx = tid; idx < total4; idx += nthreads) {
    const int row = idx >> 7;          // EMB/4 == 128 float4 per row
    const int c4  = idx & 127;
    float4 v = make_float4(0.f, 0.f, 0.f, 0.f);
    if (row < nrows) v = s4[idx];
    _Float16* d = dst + row * LDS_STRIDE + c4 * 4;
    d[0] = (_Float16)v.x; d[1] = (_Float16)v.y;
    d[2] = (_Float16)v.z; d[3] = (_Float16)v.w;
  }
}

// ---------------------------------------------------------------------------
// Kernel 1: per-caption Gram matrix  G_j = caps_j @ caps_j^T  (padded 64x64, f16)
// ---------------------------------------------------------------------------
__global__ __launch_bounds__(128) void vsa_gram_kernel(const float* __restrict__ caps,
                                                       _Float16* __restrict__ G) {
  __shared__ _Float16 caps_lds[NW_PAD * LDS_STRIDE];
  const int j   = blockIdx.x;
  const int tid = threadIdx.x;

  load_f32_rows_to_f16_lds(caps + (size_t)j * NW * EMB, caps_lds, NW, NW_PAD, tid, 128);
  __syncthreads();

  const int lane = tid & 31;
  const int wave = tid >> 5;
  const int lrow = lane & 15;
  const int lhi  = lane >> 4;
  const int n0   = wave * 16;        // each wave owns one 16-wide N column, 4 M tiles

  v8f acc[4];
#pragma unroll
  for (int m = 0; m < 4; ++m)
#pragma unroll
    for (int v = 0; v < 8; ++v) acc[m][v] = 0.0f;

  for (int k = 0; k < EMB / 32; ++k) {
    const int kb = k * 32 + lhi * 8;
    // G = C C^T: B column n is caps row n (symmetric), so B frags read rows too.
    v16h bfrag = frag_load(caps_lds, n0 + lrow, kb, LDS_STRIDE);
#pragma unroll
    for (int m = 0; m < 4; ++m) {
      v16h afrag = frag_load(caps_lds, m * 16 + lrow, kb, LDS_STRIDE);
      acc[m] = __builtin_amdgcn_wmma_f32_16x16x32_f16(false, afrag, false, bfrag,
                                                      (short)0, acc[m], false, false);
    }
  }

  _Float16* Gj = G + (size_t)j * GDIM * GDIM;
#pragma unroll
  for (int m = 0; m < 4; ++m)
#pragma unroll
    for (int v = 0; v < 8; ++v) {
      const int rr = m * 16 + lhi * 8 + v;   // C/D layout: M = 8*(lane>=16) + vgpr
      const int cc = n0 + lrow;              //             N = lane & 15
      Gj[rr * GDIM + cc] = (_Float16)acc[m][v];   // padded rows/cols are naturally 0
    }
}

// ---------------------------------------------------------------------------
// Kernel 2: fused scores -> masked softmax -> num & ||att|| -> output
// One workgroup (4 waves) per (img i, cap j) pair.
// ---------------------------------------------------------------------------
__global__ __launch_bounds__(128) void vsa_pair_kernel(const float* __restrict__ imgs,
                                                       const float* __restrict__ caps,
                                                       const long long* __restrict__ img_lens,
                                                       const long long* __restrict__ cap_lens,
                                                       const _Float16* __restrict__ G,
                                                       float* __restrict__ out) {
  extern __shared__ char smem[];
  _Float16* imgs_lds = (_Float16*)smem;                          // 48*520 halves (rows 36+ zero)
  _Float16* caps_lds = imgs_lds + NR_PAD * LDS_STRIDE;           // 64*520 halves (rows 50+ zero)
  float*    S_lds    = (float*)(caps_lds + NW_PAD * LDS_STRIDE); // 48*66 floats (scores, then P)
  _Float16* attn_lds = (_Float16*)(S_lds + NR_PAD * S_STRIDE);   // 48*72 halves
  float*    num_lds  = (float*)(attn_lds + NR_PAD * A_STRIDE);   // 48 floats

  const int j   = blockIdx.x;
  const int i   = blockIdx.y;
  const int tid = threadIdx.x;
  const int capLen = (int)cap_lens[j];
  const int imgLen = (int)img_lens[i];

  // ---- Phase 0: stage imgs_i and caps_j into LDS as zero-padded f16 tiles ----
  load_f32_rows_to_f16_lds(imgs + (size_t)i * NR * EMB, imgs_lds, NR, NR_PAD, tid, 128);
  load_f32_rows_to_f16_lds(caps + (size_t)j * NW * EMB, caps_lds, NW, NW_PAD, tid, 128);
  __syncthreads();

  const int lane = tid & 31;
  const int wave = tid >> 5;
  const int lrow = lane & 15;
  const int lhi  = lane >> 4;
  const int n0   = wave * 16;       // wave's N column (caption words n0..n0+15)

  // ---- Phase 1: S = imgs_i @ caps_j^T (raw dots), 3 M-tiles x 1 N-tile per wave ----
  {
    v8f acc[3];
#pragma unroll
    for (int m = 0; m < 3; ++m)
#pragma unroll
      for (int v = 0; v < 8; ++v) acc[m][v] = 0.0f;

    for (int k = 0; k < EMB / 32; ++k) {
      const int kb = k * 32 + lhi * 8;
      v16h bfrag = frag_load(caps_lds, n0 + lrow, kb, LDS_STRIDE);
#pragma unroll
      for (int m = 0; m < 3; ++m) {
        v16h afrag = frag_load(imgs_lds, m * 16 + lrow, kb, LDS_STRIDE);
        acc[m] = __builtin_amdgcn_wmma_f32_16x16x32_f16(false, afrag, false, bfrag,
                                                        (short)0, acc[m], false, false);
      }
    }
#pragma unroll
    for (int m = 0; m < 3; ++m)
#pragma unroll
      for (int v = 0; v < 8; ++v)
        S_lds[(m * 16 + lhi * 8 + v) * S_STRIDE + (n0 + lrow)] = acc[m][v];
  }
  __syncthreads();

  // ---- Phase 2: masked softmax per region row; num = sum_w attn*s falls out free ----
  if (tid < NR_PAD) {
    _Float16* arow = attn_lds + tid * A_STRIDE;
    if (tid < NR) {
      const float* srow = S_lds + tid * S_STRIDE;
      float mx = -1e30f;
      for (int w = 0; w < capLen; ++w) mx = fmaxf(mx, srow[w] * TEMP_INV);
      float Z = 0.0f;
      for (int w = 0; w < capLen; ++w) Z += __expf(srow[w] * TEMP_INV - mx);
      const float inv = 1.0f / Z;
      float num = 0.0f;
      for (int w = 0; w < GDIM; ++w) {
        float a = 0.0f;
        if (w < capLen) {
          const float s = srow[w];
          a = __expf(s * TEMP_INV - mx) * inv;
          num += a * s;               // num = attn . rawscores == att . img  (identity)
        }
        arow[w] = (_Float16)a;
      }
      num_lds[tid] = num;
    } else {
      for (int w = 0; w < GDIM; ++w) arow[w] = (_Float16)0.0f;  // pad rows for WMMA
      num_lds[tid] = 0.0f;
    }
  }
  __syncthreads();

  // ---- Phase 3: P = attn @ G_j  (||att||^2 = attn^T G attn), overwrite S buffer ----
  {
    const _Float16* Gj = G + (size_t)j * GDIM * GDIM;
    v8f pacc[3];
#pragma unroll
    for (int m = 0; m < 3; ++m)
#pragma unroll
      for (int v = 0; v < 8; ++v) pacc[m][v] = 0.0f;

#pragma unroll
    for (int k = 0; k < 2; ++k) {
      const int kb = k * 32 + lhi * 8;
      // G is symmetric: B column n == G row n -> contiguous global 16B reads
      union { v16h v; v8h h[2]; } ub;
      const _Float16* gp = Gj + (n0 + lrow) * GDIM + kb;
      ub.h[0] = *(const v8h*)(gp);
      ub.h[1] = *(const v8h*)(gp + 16);
#pragma unroll
      for (int m = 0; m < 3; ++m) {
        v16h afrag = frag_load(attn_lds, m * 16 + lrow, kb, A_STRIDE);
        pacc[m] = __builtin_amdgcn_wmma_f32_16x16x32_f16(false, afrag, false, ub.v,
                                                         (short)0, pacc[m], false, false);
      }
    }
#pragma unroll
    for (int m = 0; m < 3; ++m)
#pragma unroll
      for (int v = 0; v < 8; ++v)
        S_lds[(m * 16 + lhi * 8 + v) * S_STRIDE + (n0 + lrow)] = pacc[m][v];
  }
  __syncthreads();

  // ---- Phase 4: out = num / (||att|| + eps), invalid regions -> -1 ----
  if (tid < NR) {
    const _Float16* arow = attn_lds + tid * A_STRIDE;
    const float*    prow = S_lds + tid * S_STRIDE;
    float norm2 = 0.0f;
    for (int w = 0; w < NW; ++w) norm2 += (float)arow[w] * prow[w];
    norm2 = fmaxf(norm2, 0.0f);
    const float res = (tid < imgLen) ? (num_lds[tid] / (sqrtf(norm2) + EPS)) : -1.0f;
    out[((size_t)i * BC + j) * NR + tid] = res;
  }
}

extern "C" void kernel_launch(void* const* d_in, const int* in_sizes, int n_in,
                              void* d_out, int out_size, void* d_ws, size_t ws_size,
                              hipStream_t stream) {
  const float*     imgs     = (const float*)d_in[0];
  const float*     caps     = (const float*)d_in[1];
  const long long* img_lens = (const long long*)d_in[2];
  const long long* cap_lens = (const long long*)d_in[3];
  float*           out      = (float*)d_out;
  _Float16*        G        = (_Float16*)d_ws;   // 128 * 64 * 64 * 2B = 1 MB

  vsa_gram_kernel<<<dim3(BC), dim3(128), 0, stream>>>(caps, G);

  const size_t smem = (size_t)((NR_PAD + NW_PAD) * LDS_STRIDE) * sizeof(_Float16)
                    + (size_t)(NR_PAD * S_STRIDE) * sizeof(float)
                    + (size_t)(NR_PAD * A_STRIDE) * sizeof(_Float16)
                    + (size_t)NR_PAD * sizeof(float);   // ~133 KB, within 320 KB WGP LDS

  vsa_pair_kernel<<<dim3(BC, BI), dim3(128), smem, stream>>>(imgs, caps, img_lens,
                                                             cap_lens, G, out);
}
